// E2E_48979807043866
// MI455X (gfx1250) — compile-verified, matrix operand order checked by reading, not run on hardware
//
#include <hip/hip_runtime.h>
#include <hip/hip_bf16.h>

// ---------------------------------------------------------------------------
// Location-aware attention decoder for MI455X (gfx1250, wave32, WMMA).
// B=32, T=1000, E=512, D=1024, A=512, C=10, F=100, ODIM=5000, OLEN=101.
// f16 weight caches (converted once) + 4x N-blocked WMMA GEMMs.
// ---------------------------------------------------------------------------

typedef __attribute__((ext_vector_type(16))) _Float16 v16h;
typedef __attribute__((ext_vector_type(8)))  _Float16 v8h;
typedef __attribute__((ext_vector_type(8)))  float    v8f;
typedef __attribute__((ext_vector_type(2)))  float    v2f;

#define B_    32
#define T_    1000
#define E_    512
#define D_    1024
#define A_    512
#define C_    10
#define F_    100
#define ODIM_ 5000
#define L_    100
#define OLEN_ 101
#define SOS_  4998
#define KIH_  (D_ + E_)   // 1536

// ---- WMMA wrappers --------------------------------------------------------
__device__ __forceinline__ v8f wmma_f16(v16h a, v16h b, v8f c) {
  return __builtin_amdgcn_wmma_f32_16x16x32_f16(false, a, false, b, (short)0, c,
                                                false, false);
}
__device__ __forceinline__ v8f wmma_f32k4(v2f a, v2f b, v8f c) {
  return __builtin_amdgcn_wmma_f32_16x16x4_f32(false, a, false, b, (short)0, c,
                                               false, false);
}

// ---- fragment loaders -----------------------------------------------------
// A/B fragment (16 rows x 32 K, f16) from row-major fp32 (convert in regs).
// ISA 7.12.2: lane&15 = row; lanes 16..31 hold K+8; halves j<8 -> K=kh+j,
// j>=8 -> K=16+kh+(j-8).
__device__ __forceinline__ v16h load_frag_f16(const float* __restrict__ base,
                                              int ld, int lane) {
  const int m  = lane & 15;
  const int kh = (lane >> 4) << 3;
  const float* row = base + (size_t)m * ld + kh;
  v16h f;
#pragma unroll
  for (int p = 0; p < 8; ++p) {
    const int k = ((p >> 2) << 4) + ((p & 3) << 1);
    f[2 * p]     = (_Float16)row[k];
    f[2 * p + 1] = (_Float16)row[k + 1];
  }
  return f;
}

// Same fragment from a native f16 row-major matrix: two b128 loads, no cvt.
__device__ __forceinline__ v16h load_frag_h16(const _Float16* __restrict__ base,
                                              int ld, int lane) {
  const int m  = lane & 15;
  const int kh = (lane >> 4) << 3;
  const _Float16* row = base + (size_t)m * ld + kh;
  v8h lo = *(const v8h*)(row);
  v8h hi = *(const v8h*)(row + 16);
  return __builtin_shufflevector(lo, hi, 0, 1, 2, 3, 4, 5, 6, 7,
                                 8, 9, 10, 11, 12, 13, 14, 15);
}

// Row-guarded f16 fragment (for the ODIM=5000 edge of w_out).
__device__ __forceinline__ v16h load_frag_h16_guard(const _Float16* __restrict__ M0,
                                                    int nrows, int row0, int ld,
                                                    int k0, int lane) {
  const int m  = lane & 15;
  const int kh = (lane >> 4) << 3;
  const int r  = row0 + m;
  v16h f = {};
  if (r < nrows) {
    const _Float16* row = M0 + (size_t)r * ld + k0 + kh;
    v8h lo = *(const v8h*)(row);
    v8h hi = *(const v8h*)(row + 16);
    f = __builtin_shufflevector(lo, hi, 0, 1, 2, 3, 4, 5, 6, 7,
                                8, 9, 10, 11, 12, 13, 14, 15);
  }
  return f;
}

// ---- one-shot f32 -> f16 weight conversion --------------------------------
__global__ void cvt_f16_kernel(const float* __restrict__ src,
                               _Float16* __restrict__ dst, int n) {
  int i = blockIdx.x * blockDim.x + threadIdx.x;
  const int stride = gridDim.x * blockDim.x;
  for (; i < n; i += stride) dst[i] = (_Float16)src[i];
}

// ---- init: aw = 1/T, c = 0, z0 = 0 ---------------------------------------
__global__ void init_kernel(float* __restrict__ aw, float* __restrict__ cst,
                            float* __restrict__ z0) {
  const int idx = blockIdx.x * blockDim.x + threadIdx.x;   // 128*256 = 32768
  if (idx < B_ * T_) aw[idx] = 1.0f / (float)T_;
  if (idx < B_ * D_) { cst[idx] = 0.f; z0[idx] = 0.f; }
}

// ---- pre = hpad[32000,512] @ w_enc^T + b_enc  (f16 WMMA, 4x N-blocked) ---
__global__ void pre_gemm_kernel(const float* __restrict__ hpad,
                                const _Float16* __restrict__ w_enc16,
                                const float* __restrict__ b_enc,
                                float* __restrict__ pre) {
  const int wave = threadIdx.x >> 5;
  const int lane = threadIdx.x & 31;
  const int id   = blockIdx.x * 8 + wave;     // 16000 units exactly
  const int row0  = (id >> 3) * 16;           // 2000 M-tiles
  const int ncol0 = (id & 7) * 64;            // 8 groups of 4 N-tiles
  v8f acc[4] = {{}, {}, {}, {}};
  for (int k0 = 0; k0 < E_; k0 += 32) {
    __builtin_prefetch(hpad + (size_t)row0 * E_ + k0 + 64, 0, 1);
    v16h a = load_frag_f16(hpad + (size_t)row0 * E_ + k0, E_, lane);
#pragma unroll
    for (int j = 0; j < 4; ++j) {
      v16h b = load_frag_h16(w_enc16 + (size_t)(ncol0 + j * 16) * E_ + k0, E_, lane);
      acc[j] = wmma_f16(a, b, acc[j]);
    }
  }
  const int n  = lane & 15;
  const int mb = (lane >> 4) << 3;
#pragma unroll
  for (int j = 0; j < 4; ++j) {
    const int col  = ncol0 + j * 16 + n;
    const float bias = b_enc[col];
#pragma unroll
    for (int r = 0; r < 8; ++r)
      pre[(size_t)(row0 + mb + r) * A_ + col] = acc[j][r] + bias;
  }
}

// ---- dec = z[32,1024] @ w_dec^T[512,1024]  (native fp32 WMMA 16x16x4) ----
__global__ void dec_gemm_kernel(const float* __restrict__ z,
                                const float* __restrict__ w_dec,
                                float* __restrict__ dec) {
  const int wave = threadIdx.x >> 5;
  const int lane = threadIdx.x & 31;
  const int tile = blockIdx.x * 8 + wave;     // 64 tiles exactly
  const int row0 = (tile >> 5) * 16;
  const int col0 = (tile & 31) * 16;
  const int mn   = lane & 15;
  const int koff = (lane >> 4) << 1;
  const float* zr = z     + (size_t)(row0 + mn) * D_;
  const float* wr = w_dec + (size_t)(col0 + mn) * D_;
  v8f acc = {};
  for (int k0 = 0; k0 < D_; k0 += 4) {
    v2f a = { zr[k0 + koff], zr[k0 + koff + 1] };
    v2f b = { wr[k0 + koff], wr[k0 + koff + 1] };
    acc = wmma_f32k4(a, b, acc);
  }
  const int n  = lane & 15;
  const int mb = (lane >> 4) << 3;
#pragma unroll
  for (int r = 0; r < 8; ++r)
    dec[(size_t)(row0 + mb + r) * A_ + col0 + n] = acc[r];
}

// ---- attention energy: conv + tanh-MLP + gvec dot, one wave per (b,t) ----
__global__ void att_energy_kernel(const float* __restrict__ aw,
                                  const float* __restrict__ pre,
                                  const float* __restrict__ dec,
                                  const float* __restrict__ w_conv,
                                  const float* __restrict__ w_att,
                                  const float* __restrict__ w_gvec,
                                  const float* __restrict__ b_gvec,
                                  float* __restrict__ e) {
  const int b    = blockIdx.y;
  const int wave = threadIdx.x >> 5;
  const int lane = threadIdx.x & 31;
  const int t    = blockIdx.x * 8 + wave;     // grid.x = 125 -> T exactly

  float conv[C_];
#pragma unroll
  for (int c = 0; c < C_; ++c) conv[c] = 0.f;
  const float* awb = aw + (size_t)b * T_;
  for (int f = lane; f < 2 * F_ + 1; f += 32) {
    const int tt = t + f - F_;
    const float av = (tt >= 0 && tt < T_) ? awb[tt] : 0.f;
#pragma unroll
    for (int c = 0; c < C_; ++c) conv[c] += av * w_conv[c * (2 * F_ + 1) + f];
  }
#pragma unroll
  for (int c = 0; c < C_; ++c)
    for (int off = 16; off > 0; off >>= 1)
      conv[c] += __shfl_xor(conv[c], off, 32);

  const float* preb = pre + ((size_t)b * T_ + t) * A_;
  const float* decb = dec + (size_t)b * A_;
  float acc = 0.f;
  for (int a = lane; a < A_; a += 32) {
    const float* wa = w_att + a * C_;
    float af = 0.f;
#pragma unroll
    for (int c = 0; c < C_; ++c) af += conv[c] * wa[c];
    acc += w_gvec[a] * tanhf(af + preb[a] + decb[a]);
  }
  for (int off = 16; off > 0; off >>= 1) acc += __shfl_xor(acc, off, 32);
  if (lane == 0) e[(size_t)b * T_ + t] = 2.0f * (acc + b_gvec[0]); // SCALING=2
}

// ---- softmax over T + ctx = aw . hpad, one block per b -------------------
__global__ void softmax_ctx_kernel(const float* __restrict__ e,
                                   const float* __restrict__ hpad,
                                   float* __restrict__ aw,
                                   float* __restrict__ ctx) {
  const int b   = blockIdx.x;
  const int tid = threadIdx.x;                // 256 threads
  __shared__ float red[256];
  const float* eb = e + (size_t)b * T_;
  float mx = -1e30f;
  for (int t = tid; t < T_; t += 256) mx = fmaxf(mx, eb[t]);
  red[tid] = mx; __syncthreads();
  for (int s = 128; s > 0; s >>= 1) {
    if (tid < s) red[tid] = fmaxf(red[tid], red[tid + s]);
    __syncthreads();
  }
  mx = red[0]; __syncthreads();

  float sum = 0.f;
  float* awb = aw + (size_t)b * T_;
  for (int t = tid; t < T_; t += 256) {
    const float v = __expf(eb[t] - mx);
    awb[t] = v;
    sum += v;
  }
  red[tid] = sum; __syncthreads();
  for (int s = 128; s > 0; s >>= 1) {
    if (tid < s) red[tid] += red[tid + s];
    __syncthreads();
  }
  const float inv = 1.f / red[0]; __syncthreads();
  for (int t = tid; t < T_; t += 256) awb[t] *= inv;
  __syncthreads();

  const float* hb = hpad + (size_t)b * T_ * E_;
  for (int ei = tid; ei < E_; ei += 256) {
    float s0 = 0.f;
    for (int t = 0; t < T_; ++t) s0 += awb[t] * hb[(size_t)t * E_ + ei];
    ctx[(size_t)b * E_ + ei] = s0;
  }
}

// ---- LSTM gates = [embed(y);ctx] @ w_ih^T + z @ w_hh^T + biases ----------
// f16 WMMA; B operands from pre-converted f16 weights (no per-step cvt).
__global__ void lstm_gemm_kernel(const float* __restrict__ embed,
                                 const int* __restrict__ ys, int t,
                                 const float* __restrict__ ctx,
                                 const float* __restrict__ zprev,
                                 const _Float16* __restrict__ w_ih16,
                                 const _Float16* __restrict__ w_hh16,
                                 const float* __restrict__ b_ih,
                                 const float* __restrict__ b_hh,
                                 float* __restrict__ gates) {
  const int wave = threadIdx.x >> 5;
  const int lane = threadIdx.x & 31;
  const int tile = blockIdx.x * 8 + wave;     // 512 tiles exactly
  const int row0 = (tile >> 8) * 16;          // 2 M-tiles (batch)
  const int col0 = (tile & 255) * 16;         // 256 N-tiles (4096 gates)
  const int m    = lane & 15;
  const int kh   = (lane >> 4) << 3;

  const int bb   = row0 + m;
  const int yrow = (t == 0) ? SOS_ : ys[bb * L_ + (t - 1)];
  const float* erow = embed + (size_t)yrow * D_;
  const float* crow = ctx   + (size_t)bb * E_;

  v8f acc = {};
  // phase 1: x = [embed_row | ctx]  (K = 1536) vs w_ih
  for (int k0 = 0; k0 < KIH_; k0 += 32) {
    v16h a;
#pragma unroll
    for (int p = 0; p < 8; ++p) {
      const int k = k0 + kh + ((p >> 2) << 4) + ((p & 3) << 1);
      float x0, x1;
      if (k < D_) { x0 = erow[k];      x1 = erow[k + 1]; }
      else        { x0 = crow[k - D_]; x1 = crow[k - D_ + 1]; }
      a[2 * p]     = (_Float16)x0;
      a[2 * p + 1] = (_Float16)x1;
    }
    v16h bf = load_frag_h16(w_ih16 + (size_t)col0 * KIH_ + k0, KIH_, lane);
    acc = wmma_f16(a, bf, acc);
  }
  // phase 2: zprev (K = 1024) vs w_hh
  for (int k0 = 0; k0 < D_; k0 += 32) {
    __builtin_prefetch(w_hh16 + (size_t)col0 * D_ + k0 + 128, 0, 1);
    v16h a  = load_frag_f16(zprev + (size_t)row0 * D_ + k0, D_, lane);
    v16h bf = load_frag_h16(w_hh16 + (size_t)col0 * D_ + k0, D_, lane);
    acc = wmma_f16(a, bf, acc);
  }
  const int n  = lane & 15;
  const int mb = (lane >> 4) << 3;
  const float bias = b_ih[col0 + n] + b_hh[col0 + n];
#pragma unroll
  for (int r = 0; r < 8; ++r)
    gates[(size_t)(row0 + mb + r) * (4 * D_) + col0 + n] = acc[r] + bias;
}

// ---- LSTM elementwise: c,z update; z written into z_all[t] ---------------
__global__ void lstm_ew_kernel(const float* __restrict__ gates,
                               float* __restrict__ cstate,
                               float* __restrict__ zout) {
  const int idx = blockIdx.x * blockDim.x + threadIdx.x;  // 32*1024
  const int b = idx >> 10, d = idx & (D_ - 1);
  const float* g = gates + (size_t)b * (4 * D_);
  const float ig = g[d], fg = g[D_ + d], gg = g[2 * D_ + d], og = g[3 * D_ + d];
  const float si = 1.f / (1.f + __expf(-ig));
  const float sf = 1.f / (1.f + __expf(-fg));
  const float so = 1.f / (1.f + __expf(-og));
  const float cn = sf * cstate[idx] + si * tanhf(gg);
  cstate[idx] = cn;
  zout[idx]   = so * tanhf(cn);
}

// ---- output projection: [3232,1024] @ w_out^T + b_out (4x N-blocked) -----
__global__ void out_gemm_kernel(const float* __restrict__ zall,
                                const _Float16* __restrict__ w_out16,
                                const float* __restrict__ b_out,
                                float* __restrict__ out) {
  const int wave = threadIdx.x >> 5;
  const int lane = threadIdx.x & 31;
  const int NTG = 79;                          // groups of 4 N-tiles (316 >= 313)
  const int NU  = 202 * NTG;                   // 15958 wave units
  int id = blockIdx.x * 8 + wave;
  if (id >= NU) id = NU - 1;                   // wave-uniform clamp (EXEC all-1)
  const int row0  = (id / NTG) * 16;
  const int ncol0 = (id % NTG) * 64;
  v8f acc[4] = {{}, {}, {}, {}};
  for (int k0 = 0; k0 < D_; k0 += 32) {
    __builtin_prefetch(zall + (size_t)row0 * D_ + k0 + 64, 0, 1);
    v16h a = load_frag_f16(zall + (size_t)row0 * D_ + k0, D_, lane);
#pragma unroll
    for (int j = 0; j < 4; ++j) {
      v16h b = load_frag_h16_guard(w_out16, ODIM_, ncol0 + j * 16, D_, k0, lane);
      acc[j] = wmma_f16(a, b, acc[j]);
    }
  }
  const int n  = lane & 15;
  const int mb = (lane >> 4) << 3;
#pragma unroll
  for (int j = 0; j < 4; ++j) {
    const int col = ncol0 + j * 16 + n;
    if (col < ODIM_) {
      const float bias = b_out[col];
#pragma unroll
      for (int r = 0; r < 8; ++r) {
        const int row = row0 + mb + r;         // row = o*32 + b (z_all order)
        const int o = row >> 5, b = row & 31;
        out[((size_t)b * OLEN_ + o) * ODIM_ + col] = acc[j][r] + bias;
      }
    }
  }
}

// ---------------------------------------------------------------------------
extern "C" void kernel_launch(void* const* d_in, const int* in_sizes, int n_in,
                              void* d_out, int out_size, void* d_ws, size_t ws_size,
                              hipStream_t stream) {
  const float* hpad   = (const float*)d_in[0];
  const int*   ys     = (const int*)  d_in[1];
  const float* w_enc  = (const float*)d_in[2];
  const float* b_enc  = (const float*)d_in[3];
  const float* w_dec  = (const float*)d_in[4];
  const float* w_att  = (const float*)d_in[5];
  const float* w_conv = (const float*)d_in[6];
  const float* w_gvec = (const float*)d_in[7];
  const float* b_gvec = (const float*)d_in[8];
  const float* embed  = (const float*)d_in[9];
  const float* w_ih   = (const float*)d_in[10];
  const float* w_hh   = (const float*)d_in[11];
  const float* b_ih   = (const float*)d_in[12];
  const float* b_hh   = (const float*)d_in[13];
  const float* w_out  = (const float*)d_in[14];
  const float* b_out  = (const float*)d_in[15];
  float* out = (float*)d_out;

  float* ws    = (float*)d_ws;
  float* pre   = ws; ws += (size_t)B_ * T_ * A_;   // 16,384,000 f32
  float* aw    = ws; ws += B_ * T_;
  float* e     = ws; ws += B_ * T_;
  float* dec   = ws; ws += B_ * A_;
  float* ctx   = ws; ws += B_ * E_;
  float* gates = ws; ws += B_ * 4 * D_;
  float* cst   = ws; ws += B_ * D_;
  float* z0    = ws; ws += B_ * D_;
  float* zall  = ws; ws += (size_t)OLEN_ * B_ * D_; // 3,309,568 f32
  // f16 weight caches (2 halves per f32 slot)
  _Float16* w_enc16 = (_Float16*)ws; ws += (A_ * E_) / 2;          // 262,144 h
  _Float16* w_ih16  = (_Float16*)ws; ws += (4 * D_ * KIH_) / 2;    // 6,291,456 h
  _Float16* w_hh16  = (_Float16*)ws; ws += (4 * D_ * D_) / 2;      // 4,194,304 h
  _Float16* w_out16 = (_Float16*)ws; ws += (ODIM_ * D_) / 2;       // 5,120,000 h

  // one-shot weight conversion to f16
  cvt_f16_kernel<<<512, 256, 0, stream>>>(w_enc, w_enc16, A_ * E_);
  cvt_f16_kernel<<<2048, 256, 0, stream>>>(w_ih, w_ih16, 4 * D_ * KIH_);
  cvt_f16_kernel<<<2048, 256, 0, stream>>>(w_hh, w_hh16, 4 * D_ * D_);
  cvt_f16_kernel<<<2048, 256, 0, stream>>>(w_out, w_out16, ODIM_ * D_);

  init_kernel<<<128, 256, 0, stream>>>(aw, cst, z0);
  pre_gemm_kernel<<<2000, 256, 0, stream>>>(hpad, w_enc16, b_enc, pre);

  for (int t = 0; t < OLEN_; ++t) {
    const float* zprev = (t == 0) ? z0 : zall + (size_t)(t - 1) * B_ * D_;
    dec_gemm_kernel<<<8, 256, 0, stream>>>(zprev, w_dec, dec);
    att_energy_kernel<<<dim3(125, 32), 256, 0, stream>>>(
        aw, pre, dec, w_conv, w_att, w_gvec, b_gvec, e);
    softmax_ctx_kernel<<<32, 256, 0, stream>>>(e, hpad, aw, ctx);
    lstm_gemm_kernel<<<64, 256, 0, stream>>>(embed, ys, t, ctx, zprev,
                                             w_ih16, w_hh16, b_ih, b_hh, gates);
    lstm_ew_kernel<<<128, 256, 0, stream>>>(gates, cst,
                                            zall + (size_t)t * B_ * D_);
  }

  out_gemm_kernel<<<1995, 256, 0, stream>>>(zall, w_out16, b_out, out);
}
